// NeuralNetSinkhornPRModel_70308614635612
// MI455X (gfx1250) — compile-verified, matrix operand order checked by reading, not run on hardware
//
#include <hip/hip_runtime.h>
#include <hip/hip_bf16.h>
#include <cmath>
#include <vector>
#include <cstddef>

// ---------------------------------------------------------------------------
// Problem constants (from reference)
// ---------------------------------------------------------------------------
#define NPTS 2048
#define DIN  768
#define DOUT 64

typedef __attribute__((ext_vector_type(16))) _Float16 v16h;
typedef __attribute__((ext_vector_type(8)))  float    v8f;

// ---------------------------------------------------------------------------
// WMMA tile loader: 16x32 f16 operand from an f32 row-major matrix.
// Per CDNA5 ISA (CDNA5 ISA, 16-bit A-matrix 16x32):
//   lanes 0-15 : row M = lane,    elements 0..7 = K 0..7,  8..15 = K 16..23
//   lanes 16-31: row M = lane-16, elements 0..7 = K 8..15, 8..15 = K 24..31
// For D = A*B with B supplied as B^T (N x K row-major) the same pattern is the
// valid B operand, so one loader serves both sides of X @ Y^T.
// ---------------------------------------------------------------------------
__device__ inline v16h load_tile_f32_as_f16(const float* __restrict__ base,
                                            int ld, int lane) {
    int r  = lane & 15;
    int kh = (lane >> 4) << 3;            // 0 or 8
    const float* p = base + (size_t)r * ld + kh;
    v16h a;
#pragma unroll
    for (int i = 0; i < 8; ++i) a[i]     = (_Float16)p[i];
#pragma unroll
    for (int i = 0; i < 8; ++i) a[i + 8] = (_Float16)p[16 + i];
    return a;
}

// ---------------------------------------------------------------------------
// predict: out = relu(x @ W^T + bias)   x:[NPTS,DIN]  W:[DOUT,DIN]
// one wave (32 threads) per 16x16 output tile; K loop of 16x16x32 WMMAs
// ---------------------------------------------------------------------------
__global__ void __launch_bounds__(32)
predict_wmma(const float* __restrict__ x, const float* __restrict__ W,
             const float* __restrict__ bias, float* __restrict__ out) {
    const int tm = blockIdx.x, tn = blockIdx.y;
    const int lane = threadIdx.x & 31;
    v8f c = {};
    const float* xa = x + (size_t)(tm * 16) * DIN;
    const float* wb = W + (size_t)(tn * 16) * DIN;
#pragma unroll
    for (int k = 0; k < DIN; k += 32) {
        v16h a = load_tile_f32_as_f16(xa + k, DIN, lane);
        v16h b = load_tile_f32_as_f16(wb + k, DIN, lane);
        c = __builtin_amdgcn_wmma_f32_16x16x32_f16(false, a, false, b,
                                                   (short)0, c, false, false);
    }
    // C/D layout: lanes 0-15 -> N=lane, M=v ; lanes 16-31 -> N=lane-16, M=8+v
    const int ncol  = lane & 15;
    const int mbase = (lane >> 4) << 3;
    const float bv  = bias[tn * 16 + ncol];
#pragma unroll
    for (int v = 0; v < 8; ++v) {
        float val = c[v] + bv;
        out[(size_t)(tm * 16 + mbase + v) * DOUT + tn * 16 + ncol] =
            fmaxf(val, 0.0f);
    }
}

// ---------------------------------------------------------------------------
// cost: C[i,j] = sqx[i] + sqy[j] - <px_i, py_j>    (K = DOUT = 64)
// If Ct != nullptr, also writes the transpose (LDS-staged so the transposed
// stores are coalesced) so that both softmin directions can run as fully
// parallel, coalesced ROW reductions.
// ---------------------------------------------------------------------------
__global__ void __launch_bounds__(32)
cost_wmma(const float* __restrict__ px, const float* __restrict__ py,
          const float* __restrict__ sqx, const float* __restrict__ sqy,
          float* __restrict__ C, float* __restrict__ Ct) {
    __shared__ float tile[16][17];        // 16x16 + pad against bank conflicts
    const int tm = blockIdx.x, tn = blockIdx.y;
    const int lane = threadIdx.x & 31;
    v8f c = {};
    const float* xa = px + (size_t)(tm * 16) * DOUT;
    const float* yb = py + (size_t)(tn * 16) * DOUT;
#pragma unroll
    for (int k = 0; k < DOUT; k += 32) {
        v16h a = load_tile_f32_as_f16(xa + k, DOUT, lane);
        v16h b = load_tile_f32_as_f16(yb + k, DOUT, lane);
        c = __builtin_amdgcn_wmma_f32_16x16x32_f16(false, a, false, b,
                                                   (short)0, c, false, false);
    }
    const int ncol  = lane & 15;
    const int mbase = (lane >> 4) << 3;
    const float sy  = sqy[tn * 16 + ncol];
#pragma unroll
    for (int v = 0; v < 8; ++v) {
        int m = tm * 16 + mbase + v;
        float val = sqx[m] + sy - c[v];
        C[(size_t)m * NPTS + tn * 16 + ncol] = val;
        tile[mbase + v][ncol] = val;
    }
    if (Ct) {                              // uniform branch across the block
        __syncthreads();
        const int r = lane & 15;           // Ct tile row == C tile column
        const int h = (lane >> 4) << 3;    // which half of the Ct row
#pragma unroll
        for (int i = 0; i < 8; ++i) {
            Ct[(size_t)(tn * 16 + r) * NPTS + tm * 16 + h + i] =
                tile[h + i][r];
        }
    }
}

// ---------------------------------------------------------------------------
// sq[i] = 0.5 * ||p_i||^2
// ---------------------------------------------------------------------------
__global__ void row_halfsq(const float* __restrict__ p, float* __restrict__ sq) {
    int r = blockIdx.x * blockDim.x + threadIdx.x;
    if (r >= NPTS) return;
    float s = 0.f;
    const float* row = p + (size_t)r * DOUT;
#pragma unroll 8
    for (int k = 0; k < DOUT; ++k) { float v = row[k]; s += v * v; }
    sq[r] = 0.5f * s;
}

__global__ void zero_vec(float* __restrict__ v, int n) {
    int i = blockIdx.x * blockDim.x + threadIdx.x;
    if (i < n) v[i] = 0.0f;
}

// ---------------------------------------------------------------------------
// softmin over rows of M (either C or C^T):
//   ft_i = -eps * ( logw + LSE_j (g_j - M_ij)/eps )
// one 256-thread block per row; per-thread online LSE over an unrolled
// NPTS/256 = 8 element strip (coalesced), then block combine.
// avg!=0 -> out = 0.5*(f_in + ft), else out = ft.
// ---------------------------------------------------------------------------
__global__ void __launch_bounds__(256)
softmin_rows(const float* __restrict__ M, const float* __restrict__ g,
             const float* __restrict__ f_in, float* __restrict__ f_out,
             float logw, float eps, int avg) {
    const int row = blockIdx.x;
    const float* Mrow = M + (size_t)row * NPTS;
    const float inv_eps = 1.0f / eps;

    float m = -INFINITY, s = 0.0f;
#pragma unroll
    for (int jj = 0; jj < NPTS / 256; ++jj) {
        int j = threadIdx.x + jj * 256;
        float z = (g[j] - Mrow[j]) * inv_eps;
        if (z > m) { s = s * __expf(m - z) + 1.0f; m = z; }
        else       { s += __expf(z - m); }
    }
    __shared__ float sm[256], ss[256];
    sm[threadIdx.x] = m; ss[threadIdx.x] = s;
    __syncthreads();
    for (int off = 128; off > 0; off >>= 1) {
        if ((int)threadIdx.x < off) {
            float m1 = sm[threadIdx.x], s1 = ss[threadIdx.x];
            float m2 = sm[threadIdx.x + off], s2 = ss[threadIdx.x + off];
            float mm = fmaxf(m1, m2);
            float sn = 0.0f;
            if (m1 != -INFINITY) sn += s1 * __expf(m1 - mm);
            if (m2 != -INFINITY) sn += s2 * __expf(m2 - mm);
            sm[threadIdx.x] = mm; ss[threadIdx.x] = sn;
        }
        __syncthreads();
    }
    if (threadIdx.x == 0) {
        float ft = -eps * (sm[0] + __logf(ss[0]) + logw);
        f_out[row] = avg ? 0.5f * (f_in[row] + ft) : ft;
    }
}

// ---------------------------------------------------------------------------
// acc[0] = scale * sum_i ( (u1-u2)[i] + (v1-v2)[i] )   (single block)
// ---------------------------------------------------------------------------
__global__ void __launch_bounds__(256)
acc_pair(const float* __restrict__ u1, const float* __restrict__ u2,
         const float* __restrict__ v1, const float* __restrict__ v2,
         float scale, float* __restrict__ acc) {
    float s = 0.0f;
    for (int i = threadIdx.x; i < NPTS; i += blockDim.x)
        s += (u1[i] - u2[i]) + (v1[i] - v2[i]);
    __shared__ float sh[256];
    sh[threadIdx.x] = s;
    __syncthreads();
    for (int off = 128; off > 0; off >>= 1) {
        if ((int)threadIdx.x < off) sh[threadIdx.x] += sh[threadIdx.x + off];
        __syncthreads();
    }
    if (threadIdx.x == 0) acc[0] = scale * sh[0];
}

__global__ void finalize_sigmoid(const float* __restrict__ acc,
                                 float* __restrict__ out, float k) {
    if (threadIdx.x == 0 && blockIdx.x == 0) {
        float d = k * (acc[1] - acc[0]);
        out[0] = 1.0f / (1.0f + __expf(-d));
    }
}

// ---------------------------------------------------------------------------
// host orchestration
// ---------------------------------------------------------------------------
extern "C" void kernel_launch(void* const* d_in, const int* /*in_sizes*/,
                              int /*n_in*/, void* d_out, int /*out_size*/,
                              void* d_ws, size_t /*ws_size*/,
                              hipStream_t stream) {
    const float* xd = (const float*)d_in[0];
    const float* s1 = (const float*)d_in[1];
    const float* s2 = (const float*)d_in[2];
    const float* W  = (const float*)d_in[3];
    const float* bb = (const float*)d_in[4];
    float* out = (float*)d_out;

    // ---- workspace carve (all fp32) ----
    char* ws = (char*)d_ws;
    auto take = [&](size_t nfloats) {
        float* p = (float*)ws;
        ws += nfloats * sizeof(float);
        return p;
    };
    float* C  = take((size_t)NPTS * NPTS);   // 16 MB, reused per sub-problem
    float* Ct = take((size_t)NPTS * NPTS);   // 16 MB transpose (xy problems)
    float* P[3];  for (int i = 0; i < 3; ++i) P[i]  = take((size_t)NPTS * DOUT);
    float* SQ[3]; for (int i = 0; i < 3; ++i) SQ[i] = take(NPTS);
    float* f0 = take(NPTS); float* f1 = take(NPTS);
    float* g0 = take(NPTS); float* g1 = take(NPTS);
    float* fxy = take(NPTS); float* gxy = take(NPTS);
    float* fxx = take(NPTS); float* gyy = take(NPTS);
    float* acc = take(2);

    // ---- epsilon schedule (geomloss-style), computed on host ----
    const double eps0 = 32.0, eps_f = 0.05 * 0.05, ratio = 0.9 * 0.9;
    int nsteps = (int)std::ceil(std::log(eps_f / eps0) / std::log(ratio));
    std::vector<float> sched;
    for (int k = 0; k <= nsteps; ++k) {
        double e = eps0 * std::pow(ratio, (double)k);
        sched.push_back((float)(e > eps_f ? e : eps_f));
    }
    sched.push_back((float)eps_f);               // 47 entries total
    const float epsF = (float)eps_f;
    const float logw = -std::log((float)NPTS);   // log a_i == log b_j (uniform)

    const dim3 wave(32);
    const dim3 pgrid(NPTS / 16, DOUT / 16);      // predict tiles
    const dim3 cgrid(NPTS / 16, NPTS / 16);      // cost tiles
    const dim3 b256(256);
    const dim3 rows(NPTS);
    const dim3 vgrid(NPTS / 256);

    auto zero = [&](float* v) {
        zero_vec<<<vgrid, b256, 0, stream>>>(v, NPTS);
    };

    // symmetric OT problem on C (C is symmetric, no transpose needed);
    // result potential -> fout
    auto run_xx = [&](float* fout) {
        zero(f0);
        float *fa = f0, *fb = f1;
        for (float e : sched) {
            softmin_rows<<<rows, b256, 0, stream>>>(C, fa, fa, fb, logw, e, 1);
            float* t = fa; fa = fb; fb = t;
        }
        softmin_rows<<<rows, b256, 0, stream>>>(C, fa, fa, fout, logw, epsF, 0);
    };

    // asymmetric OT problem on (C, Ct); potentials -> (fo, go).
    // g-updates are row-softmins over Ct: full 2048-block parallelism,
    // fully coalesced, both matrices L2-resident across all iterations.
    auto run_xy = [&](float* fo, float* go) {
        zero(f0); zero(g0);
        float *fa = f0, *fb = f1, *ga = g0, *gb = g1;
        for (float e : sched) {
            softmin_rows<<<rows, b256, 0, stream>>>(C,  ga, fa, fb, logw, e, 1);
            softmin_rows<<<rows, b256, 0, stream>>>(Ct, fa, ga, gb, logw, e, 1);
            float* t;
            t = fa; fa = fb; fb = t;
            t = ga; ga = gb; gb = t;
        }
        softmin_rows<<<rows, b256, 0, stream>>>(C,  ga, fa, fo, logw, epsF, 0);
        softmin_rows<<<rows, b256, 0, stream>>>(Ct, fa, ga, go, logw, epsF, 0);
    };

    // ---- predict all three point clouds ----
    const float* X[3] = { xd, s1, s2 };
    for (int i = 0; i < 3; ++i) {
        predict_wmma<<<pgrid, wave, 0, stream>>>(X[i], W, bb, P[i]);
        row_halfsq<<<vgrid, b256, 0, stream>>>(P[i], SQ[i]);
    }

    const float scale = 1.0f / (float)NPTS;

    // ---- divergence 1: S(pd, p1) ----
    cost_wmma<<<cgrid, wave, 0, stream>>>(P[0], P[1], SQ[0], SQ[1], C, Ct);
    run_xy(fxy, gxy);
    cost_wmma<<<cgrid, wave, 0, stream>>>(P[0], P[0], SQ[0], SQ[0], C, nullptr);
    run_xx(fxx);                                   // kept for divergence 2 too
    cost_wmma<<<cgrid, wave, 0, stream>>>(P[1], P[1], SQ[1], SQ[1], C, nullptr);
    run_xx(gyy);
    acc_pair<<<1, b256, 0, stream>>>(fxy, fxx, gxy, gyy, scale, acc + 0);

    // ---- divergence 2: S(pd, p2) (reuses fxx) ----
    cost_wmma<<<cgrid, wave, 0, stream>>>(P[0], P[2], SQ[0], SQ[2], C, Ct);
    run_xy(fxy, gxy);
    cost_wmma<<<cgrid, wave, 0, stream>>>(P[2], P[2], SQ[2], SQ[2], C, nullptr);
    run_xx(gyy);
    acc_pair<<<1, b256, 0, stream>>>(fxy, fxx, gxy, gyy, scale, acc + 1);

    // ---- sigmoid(SCALING_FACTOR * (dist2 - dist1)) ----
    finalize_sigmoid<<<1, 32, 0, stream>>>(acc, out, 10.0f);
}
